// GAT_raw_att_12_68968584839884
// MI455X (gfx1250) — matrix operand order repaired and checked by reading, not compiled, not run on hardware
//
#include <hip/hip_runtime.h>

#define NND 8192
#define NE  131072
#define NG  32
#define SCALE_ATT 0.044194173824159216f   // 1/sqrt(512)

typedef __attribute__((ext_vector_type(16))) __bf16 v16bf;
typedef __attribute__((ext_vector_type(8)))  float  v8f;

__device__ inline unsigned short f2bf(float f){
  unsigned u = __float_as_uint(f);
  u += 0x7fffu + ((u >> 16) & 1u);          // round-to-nearest-even
  return (unsigned short)(u >> 16);
}
__device__ inline unsigned pack2bf(float a, float b){
  return (unsigned)f2bf(a) | ((unsigned)f2bf(b) << 16);
}

// ---------------------------------------------------------------------------
// Universal batched WMMA GEMM:
//   C[z] = act( alpha * A[z] (x) B[z] + bias ),  B transposed if TB.
// Tile: 128(M) x 128(N) x 32(K); 8 waves, wave w -> 16 cols, 8 M-subtiles.
// Software pipelined: tile k+1 global loads issued between LDS publish and
// the 64 WMMAs of tile k. FULL=1 instances have no bounds guards.
// LDS holds bf16 packed as (k,k+1) dword pairs in the ISA 7.12.2 layouts.
// ---------------------------------------------------------------------------
template<int FULL, int TB>
__global__ __launch_bounds__(256) void gemm_wmma_kernel(
    const float* __restrict__ A, const float* __restrict__ B,
    const float* __restrict__ bias, float* __restrict__ C,
    int M, int N, int K, int lda, int ldb, int ldc,
    long sAb, long sAh, long sBb, long sBh, long sCb, long sCh,
    int nbh, float alpha, int act)
{
  __shared__ unsigned ldsA[128 * 16];   // [row][kpair]
  __shared__ unsigned ldsB[128 * 16];   // [col][kpair]
  int zb = blockIdx.z / nbh, zh = blockIdx.z % nbh;
  A += (long)zb * sAb + (long)zh * sAh;
  B += (long)zb * sBb + (long)zh * sBh;
  C += (long)zb * sCb + (long)zh * sCh;
  int m0 = blockIdx.x * 128;
  int n0 = blockIdx.y * 128;
  int tid = threadIdx.x;
  int wave = tid >> 5, lane = tid & 31;
  int lj = lane & 15, hi = lane >> 4;

  v8f acc[8] = {};
  float aa[8], ab[8], ba[8], bb_[8];

  auto loadTiles = [&](int kb){
    #pragma unroll
    for (int it = 0; it < 8; it++){
      int i = tid + it * 256;
      int r = i >> 4, p = i & 15;
      int grow = m0 + r, gk = kb + p * 2;
      if (FULL){
        aa[it] = A[(long)grow * lda + gk];
        ab[it] = A[(long)grow * lda + gk + 1];
      } else {
        aa[it] = (grow < M && gk     < K) ? A[(long)grow * lda + gk    ] : 0.f;
        ab[it] = (grow < M && gk + 1 < K) ? A[(long)grow * lda + gk + 1] : 0.f;
      }
    }
    #pragma unroll
    for (int it = 0; it < 8; it++){
      int i = tid + it * 256;
      if (!TB){
        int p = i >> 7, colv = i & 127;
        int gcol = n0 + colv, gk = kb + p * 2;
        if (FULL){
          ba[it]  = B[(long)gk * ldb + gcol];
          bb_[it] = B[(long)(gk + 1) * ldb + gcol];
        } else {
          ba[it]  = (gcol < N && gk     < K) ? B[(long)gk * ldb + gcol]       : 0.f;
          bb_[it] = (gcol < N && gk + 1 < K) ? B[(long)(gk + 1) * ldb + gcol] : 0.f;
        }
      } else {
        int colv = i >> 4, p = i & 15;
        int gn = n0 + colv, gk = kb + p * 2;
        if (FULL){
          ba[it]  = B[(long)gn * ldb + gk];
          bb_[it] = B[(long)gn * ldb + gk + 1];
        } else {
          ba[it]  = (gn < N && gk     < K) ? B[(long)gn * ldb + gk    ] : 0.f;
          bb_[it] = (gn < N && gk + 1 < K) ? B[(long)gn * ldb + gk + 1] : 0.f;
        }
      }
    }
  };

  loadTiles(0);
  for (int kb = 0; kb < K; kb += 32){
    if (kb) __syncthreads();            // previous tile fully consumed
    #pragma unroll
    for (int it = 0; it < 8; it++){
      int i = tid + it * 256;
      ldsA[i] = pack2bf(aa[it], ab[it]);
    }
    #pragma unroll
    for (int it = 0; it < 8; it++){
      int i = tid + it * 256;
      int colv = TB ? (i >> 4) : (i & 127);
      int p    = TB ? (i & 15) : (i >> 7);
      ldsB[colv * 16 + p] = pack2bf(ba[it], bb_[it]);
    }
    __syncthreads();
    if (kb + 32 < K) loadTiles(kb + 32);   // overlap next loads with WMMAs

    union { unsigned u[8]; v16bf v; } afr, bfr;
    #pragma unroll
    for (int j = 0; j < 8; j++)            // B: interleaved {0..7,16..23}/{+8}
      bfr.u[j] = ldsB[(wave * 16 + lj) * 16 + (j & 3) + ((j & 4) << 1) + (hi << 2)];
    #pragma unroll
    for (int ms = 0; ms < 8; ms++){
      int row = ms * 16 + lj;
      #pragma unroll
      for (int j = 0; j < 8; j++)          // A: interleaved {0..7,16..23}/{+8}
        afr.u[j] = ldsA[row * 16 + (j & 3) + ((j & 4) << 1) + (hi << 2)];
      acc[ms] = __builtin_amdgcn_wmma_f32_16x16x32_bf16(
          false, afr.v, false, bfr.v, (short)0, acc[ms], false, false);
    }
  }

  int col = n0 + wave * 16 + lj;
  if (FULL || col < N){
    float bv = bias ? bias[col] : 0.f;
    #pragma unroll
    for (int ms = 0; ms < 8; ms++)
      #pragma unroll
      for (int r = 0; r < 8; r++){
        int row = m0 + ms * 16 + r + hi * 8;
        if (FULL || row < M){
          float v = acc[ms][r] * alpha + bv;
          if (act == 1 && v < 0.f) v = 0.f;
          C[(long)row * ldc + col] = v;
        }
      }
  }
}

// ------------------------- CSR build -------------------------
__global__ void count_kernel(const int* __restrict__ dst, const float* __restrict__ ea,
                             int* cnt, float* loopea){
  int e = blockIdx.x * 256 + threadIdx.x; if (e >= NE) return;
  int d = dst[e];
  atomicAdd(&cnt[d], 1);
  for (int k = 0; k < 6; k++) atomicAdd(&loopea[d * 6 + k], ea[e * 6 + k]);
}
__global__ __launch_bounds__(256) void scan_kernel(const int* __restrict__ cnt, int* indptr){
  __shared__ int part[256];
  int t = threadIdx.x;
  int s = 0;
  for (int i = 0; i < 32; i++) s += cnt[t * 32 + i];
  part[t] = s; __syncthreads();
  if (t == 0) { int run = 0; for (int i = 0; i < 256; i++){ int tmp = part[i]; part[i] = run; run += tmp; } }
  __syncthreads();
  int run = part[t];
  for (int i = 0; i < 32; i++){ indptr[t * 32 + i] = run; run += cnt[t * 32 + i]; }
  if (t == 255) indptr[NND] = run;
}
__global__ void scatter_kernel(const int* __restrict__ dst, const int* __restrict__ indptr,
                               int* fill, int* csr){
  int e = blockIdx.x * 256 + threadIdx.x; if (e >= NE) return;
  int d = dst[e];
  int p = atomicAdd(&fill[d], 1);
  csr[indptr[d] + p] = e;
}
__global__ void loopdiv_kernel(const int* __restrict__ cnt, float* loopea){
  int n = blockIdx.x * 256 + threadIdx.x; if (n >= NND) return;
  float inv = 1.f / fmaxf((float)cnt[n], 1.f);
  for (int k = 0; k < 6; k++) loopea[n * 6 + k] *= inv;
}

// ------------------------- GAT pieces -------------------------
// we_red[k][h] = sum_c We[k, h*c + c_i] * att_edge[h, c_i]   (6x4)
__global__ void wered_kernel(const float* __restrict__ We, const float* __restrict__ ae,
                             float* wr, int c){
  int t = threadIdx.x; if (t >= 24) return;
  int k = t / 4, hd = t % 4, HC = 4 * c;
  float s = 0.f;
  for (int ci = 0; ci < c; ci++) s += We[k * HC + hd * c + ci] * ae[hd * c + ci];
  wr[k * 4 + hd] = s;
}
// per-node attention scalars: a_src, a_dst, self-loop edge term (one wave/head)
__global__ __launch_bounds__(128) void attnode_kernel(
    const float* __restrict__ h, const float* __restrict__ att_src,
    const float* __restrict__ att_dst, const float* __restrict__ loopea,
    const float* __restrict__ wr, float* a_src, float* a_dst, float* a_self, int c){
  int n = blockIdx.x;
  int hd = threadIdx.x >> 5, lane = threadIdx.x & 31;
  int HC = 4 * c;
  const float* hr = h + (long)n * HC + hd * c;
  float s = 0.f, d = 0.f;
  for (int ci = lane; ci < c; ci += 32){
    float hv = hr[ci];
    s += hv * att_src[hd * c + ci];
    d += hv * att_dst[hd * c + ci];
  }
  for (int off = 16; off; off >>= 1){ s += __shfl_xor(s, off, 32); d += __shfl_xor(d, off, 32); }
  if (lane == 0){
    a_src[n * 4 + hd] = s; a_dst[n * 4 + hd] = d;
    float e = 0.f;
    for (int k = 0; k < 6; k++) e += loopea[n * 6 + k] * wr[k * 4 + hd];
    a_self[n * 4 + hd] = e;
  }
}
__global__ void alphaedge_kernel(const float* __restrict__ ea, const float* __restrict__ wr,
                                 float* aE){
  int idx = blockIdx.x * 256 + threadIdx.x; if (idx >= NE * 4) return;
  int e = idx >> 2, hd = idx & 3;
  float s = 0.f;
  for (int k = 0; k < 6; k++) s += ea[e * 6 + k] * wr[k * 4 + hd];
  aE[idx] = s;
}
// segment softmax over incoming edges + self loop (block per dst node)
__global__ __launch_bounds__(256) void gatsm_kernel(
    const int* __restrict__ indptr, const int* __restrict__ eid,
    const int* __restrict__ srcA, const float* __restrict__ a_src,
    const float* __restrict__ a_dst, const float* __restrict__ a_self,
    float* __restrict__ aE /* in: raw edge term, out: normalized w */,
    float* __restrict__ w_self){
  __shared__ float red[256];
  __shared__ float mh[4], sh[4];
  int n = blockIdx.x;
  int g = threadIdx.x >> 6, l = threadIdx.x & 63;
  int s0 = indptr[n], s1 = indptr[n + 1];
  float ad = a_dst[n * 4 + g];
  float asf = a_src[n * 4 + g] + ad + a_self[n * 4 + g];
  asf = asf > 0.f ? asf : 0.2f * asf;
  float mx = (l == 0) ? asf : -1e30f;
  for (int i = s0 + l; i < s1; i += 64){
    int e = eid[i]; int s = srcA[e];
    float a = a_src[s * 4 + g] + ad + aE[e * 4 + g];
    a = a > 0.f ? a : 0.2f * a;
    mx = fmaxf(mx, a);
  }
  red[threadIdx.x] = mx; __syncthreads();
  for (int off = 32; off; off >>= 1){
    if (l < off) red[threadIdx.x] = fmaxf(red[threadIdx.x], red[threadIdx.x + off]);
    __syncthreads();
  }
  if (l == 0) mh[g] = red[g * 64];
  __syncthreads();
  float m = mh[g];
  float sm = (l == 0) ? __expf(asf - m) : 0.f;
  for (int i = s0 + l; i < s1; i += 64){
    int e = eid[i]; int s = srcA[e];
    float a = a_src[s * 4 + g] + ad + aE[e * 4 + g];
    a = a > 0.f ? a : 0.2f * a;
    sm += __expf(a - m);
  }
  red[threadIdx.x] = sm; __syncthreads();
  for (int off = 32; off; off >>= 1){
    if (l < off) red[threadIdx.x] += red[threadIdx.x + off];
    __syncthreads();
  }
  if (l == 0) sh[g] = red[g * 64];
  __syncthreads();
  float inv = 1.f / sh[g];
  if (l == 0) w_self[n * 4 + g] = __expf(asf - m) * inv;
  for (int i = s0 + l; i < s1; i += 64){
    int e = eid[i]; int s = srcA[e];
    float a = a_src[s * 4 + g] + ad + aE[e * 4 + g];
    a = a > 0.f ? a : 0.2f * a;
    aE[e * 4 + g] = __expf(a - m) * inv;
  }
}
// weighted gather + bias + BN(eval) + ELU, edge chunks staged in LDS
__global__ __launch_bounds__(256) void gatagg_kernel(
    const float* __restrict__ h, const int* __restrict__ indptr,
    const int* __restrict__ eid, const int* __restrict__ srcA,
    const float* __restrict__ wE, const float* __restrict__ w_self,
    const float* __restrict__ bias,
    const float* __restrict__ bg, const float* __restrict__ bb,
    const float* __restrict__ bm, const float* __restrict__ bvv,
    float* __restrict__ out, int c){
  __shared__ int   s_src[128];
  __shared__ float s_w[128 * 4];
  int n = blockIdx.x, tid = threadIdx.x;
  int HC = 4 * c;
  int s0 = indptr[n], s1 = indptr[n + 1];
  int nv = (HC + 255) / 256;
  float acc[8]; int hdv[8];
  for (int v = 0; v < nv; v++){
    int oc = tid + v * 256;
    if (oc < HC){ int hd = oc / c; hdv[v] = hd; acc[v] = w_self[n * 4 + hd] * h[(long)n * HC + oc]; }
    else { hdv[v] = 0; acc[v] = 0.f; }
  }
  for (int base = s0; base < s1; base += 128){
    int cc = s1 - base; if (cc > 128) cc = 128;
    __syncthreads();
    if (tid < cc){
      int e = eid[base + tid];
      s_src[tid] = srcA[e];
      s_w[tid * 4 + 0] = wE[e * 4 + 0]; s_w[tid * 4 + 1] = wE[e * 4 + 1];
      s_w[tid * 4 + 2] = wE[e * 4 + 2]; s_w[tid * 4 + 3] = wE[e * 4 + 3];
    }
    __syncthreads();
    for (int j = 0; j < cc; j++){
      long srow = (long)s_src[j] * HC;
      for (int v = 0; v < nv; v++){
        int oc = tid + v * 256;
        if (oc < HC) acc[v] += s_w[j * 4 + hdv[v]] * h[srow + oc];
      }
    }
  }
  for (int v = 0; v < nv; v++){
    int oc = tid + v * 256;
    if (oc < HC){
      float val = acc[v] + bias[oc];
      val = bg[oc] * (val - bm[oc]) * rsqrtf(bvv[oc] + 1e-5f) + bb[oc];
      out[(long)n * HC + oc] = val > 0.f ? val : expm1f(val);
    }
  }
}

// ------------------------- set-transformer pieces -------------------------
__global__ __launch_bounds__(256) void lnres_kernel(
    const float* __restrict__ x, const float* __restrict__ res, long resStride,
    const float* __restrict__ g, const float* __restrict__ b, float* __restrict__ out){
  __shared__ float red[256];
  __shared__ float stat[2];
  int row = blockIdx.x, tid = threadIdx.x;
  const float* xr = x + (long)row * 2048;
  const float* rr = res + (long)row * resStride;
  float v[8]; float s = 0.f;
  for (int i = 0; i < 8; i++){ int ci = tid + i * 256; v[i] = xr[ci] + rr[ci]; s += v[i]; }
  red[tid] = s; __syncthreads();
  for (int off = 128; off; off >>= 1){ if (tid < off) red[tid] += red[tid + off]; __syncthreads(); }
  if (tid == 0) stat[0] = red[0] * (1.f / 2048.f);
  __syncthreads();
  float mu = stat[0]; s = 0.f;
  for (int i = 0; i < 8; i++){ float d = v[i] - mu; s += d * d; }
  red[tid] = s; __syncthreads();
  for (int off = 128; off; off >>= 1){ if (tid < off) red[tid] += red[tid + off]; __syncthreads(); }
  if (tid == 0) stat[1] = rsqrtf(red[0] * (1.f / 2048.f) + 1e-5f);
  __syncthreads();
  float inv = stat[1];
  for (int i = 0; i < 8; i++){
    int ci = tid + i * 256;
    out[(long)row * 2048 + ci] = g[ci] * (v[i] - mu) * inv + b[ci];
  }
}
__global__ __launch_bounds__(256) void softmax256_kernel(float* __restrict__ s){
  __shared__ float red[256];
  __shared__ float st;
  long row = blockIdx.x; float* r = s + row * 256; int tid = threadIdx.x;
  float v = r[tid];
  red[tid] = v; __syncthreads();
  for (int off = 128; off; off >>= 1){ if (tid < off) red[tid] = fmaxf(red[tid], red[tid + off]); __syncthreads(); }
  if (tid == 0) st = red[0];
  __syncthreads();
  float e = __expf(v - st);
  red[tid] = e; __syncthreads();
  for (int off = 128; off; off >>= 1){ if (tid < off) red[tid] += red[tid + off]; __syncthreads(); }
  if (tid == 0) st = red[0];
  __syncthreads();
  r[tid] = e / st;
}
__global__ void bnelu_kernel(float* x, const float* g, const float* b,
                             const float* m, const float* v, int total, int d){
  int i = blockIdx.x * 256 + threadIdx.x; if (i >= total) return;
  int col = i % d;
  float val = g[col] * (x[i] - m[col]) * rsqrtf(v[col] + 1e-5f) + b[col];
  x[i] = val > 0.f ? val : expm1f(val);
}
__global__ void nanfix_kernel(float* x, int n){
  int i = blockIdx.x * 256 + threadIdx.x; if (i >= n) return;
  float v = x[i];
  if (!(v == v)) v = 0.f;
  else if (v >  3.4028234663852886e38f) v =  3.4028234663852886e38f;
  else if (v < -3.4028234663852886e38f) v = -3.4028234663852886e38f;
  x[i] = v;
}
__global__ void logsm2_kernel(const float* __restrict__ z, float* __restrict__ out){
  int t = threadIdx.x; if (t >= NG) return;
  float v0 = z[t * 2], v1 = z[t * 2 + 1];
  float m = fmaxf(v0, v1);
  float l = m + logf(__expf(v0 - m) + __expf(v1 - m));
  out[t * 2] = v0 - l; out[t * 2 + 1] = v1 - l;
}

// ===========================================================================
extern "C" void kernel_launch(void* const* d_in, const int* in_sizes, int n_in,
                              void* d_out, int out_size, void* d_ws, size_t ws_size,
                              hipStream_t stream) {
  (void)in_sizes; (void)n_in; (void)out_size; (void)ws_size;
  int ii = 0;
  const float* x  = (const float*)d_in[ii++];          // 0
  const int*   ei = (const int*)  d_in[ii++];          // 1  edge_index [2,E]
  const float* ea = (const float*)d_in[ii++];          // 2  edge_feature [E,6]
  ii++;                                                // 3  batch (implied)
  const float *gW[5], *gWe[5], *gAs[5], *gAd[5], *gAe[5], *gB[5];
  for (int l = 0; l < 5; l++){
    gW[l]  = (const float*)d_in[ii++]; gWe[l] = (const float*)d_in[ii++];
    gAs[l] = (const float*)d_in[ii++]; gAd[l] = (const float*)d_in[ii++];
    gAe[l] = (const float*)d_in[ii++]; gB[l]  = (const float*)d_in[ii++];
  }
  const float *bnG[8], *bnB[8], *bnM[8], *bnV[8];
  for (int l = 0; l < 8; l++){
    bnG[l] = (const float*)d_in[ii++]; bnB[l] = (const float*)d_in[ii++];
    bnM[l] = (const float*)d_in[ii++]; bnV[l] = (const float*)d_in[ii++];
  }
  const float* enc[14]; const float* pma[14]; const float* dec[14];
  for (int j = 0; j < 14; j++) enc[j] = (const float*)d_in[ii++];
  for (int j = 0; j < 14; j++) pma[j] = (const float*)d_in[ii++];
  for (int j = 0; j < 14; j++) dec[j] = (const float*)d_in[ii++];
  const float* pmaW = (const float*)d_in[ii++];
  const float* pmab = (const float*)d_in[ii++];
  const float* seed = (const float*)d_in[ii++];
  const float *fcW[4], *fcb[4];
  for (int j = 0; j < 4; j++){ fcW[j] = (const float*)d_in[ii++]; fcb[j] = (const float*)d_in[ii++]; }
  // mab order: 0 Wq 1 bq 2 Wk 3 bk 4 Wv 5 bv 6 Wo 7 bo 8 Wl 9 bl 10 g1 11 b1 12 g2 13 b2

  // -------- workspace carve --------
  size_t off = 0;
  auto carve = [&](size_t bytes){ void* p = (char*)d_ws + off; off += (bytes + 255) & ~(size_t)255; return p; };
  const size_t BIG = (size_t)NND * 2048 * 4;
  float* bufX = (float*)carve(BIG);
  float* bufY = (float*)carve(BIG);
  float* bufP = (float*)carve(BIG);
  float* bufQ = (float*)carve(BIG);
  float* bufK = (float*)carve(BIG);
  float* bufV = (float*)carve(BIG);
  float* bufS = (float*)carve((size_t)NG * 4 * 256 * 256 * 4);
  int*   cnt    = (int*)  carve(NND * 4);
  int*   indptr = (int*)  carve((NND + 1) * 4);
  int*   fill   = (int*)  carve(NND * 4);
  int*   csr    = (int*)  carve(NE * 4);
  float* loopea = (float*)carve(NND * 6 * 4);
  float* a_src  = (float*)carve(NND * 4 * 4);
  float* a_dst  = (float*)carve(NND * 4 * 4);
  float* a_self = (float*)carve(NND * 4 * 4);
  float* alphaE = (float*)carve((size_t)NE * 4 * 4);
  float* w_self = (float*)carve(NND * 4 * 4);
  float* wered  = (float*)carve(24 * 4);
  float* qseed  = (float*)carve(2048 * 4);
  float* spma   = (float*)carve(128 * 256 * 4);

  const int* srcA = ei;
  const int* dstA = ei + NE;

  auto gemm = [&](const float* A, const float* B, const float* bias, float* C,
                  int M, int Nc, int K, int lda, int ldb, int ldc,
                  long sAb, long sAh, long sBb, long sBh, long sCb, long sCh,
                  int nbh, int nz, float alpha, int act, int transB){
    dim3 g((M + 127) / 128, (Nc + 127) / 128, nz);
    bool full = (M % 128 == 0) && (Nc % 128 == 0) && (K % 32 == 0);
    if (full) {
      if (!transB) gemm_wmma_kernel<1,0><<<g, 256, 0, stream>>>(A, B, bias, C, M, Nc, K,
          lda, ldb, ldc, sAb, sAh, sBb, sBh, sCb, sCh, nbh, alpha, act);
      else         gemm_wmma_kernel<1,1><<<g, 256, 0, stream>>>(A, B, bias, C, M, Nc, K,
          lda, ldb, ldc, sAb, sAh, sBb, sBh, sCb, sCh, nbh, alpha, act);
    } else {
      if (!transB) gemm_wmma_kernel<0,0><<<g, 256, 0, stream>>>(A, B, bias, C, M, Nc, K,
          lda, ldb, ldc, sAb, sAh, sBb, sBh, sCb, sCh, nbh, alpha, act);
      else         gemm_wmma_kernel<0,1><<<g, 256, 0, stream>>>(A, B, bias, C, M, Nc, K,
          lda, ldb, ldc, sAb, sAh, sBb, sBh, sCb, sCh, nbh, alpha, act);
    }
  };
  auto gemmP = [&](const float* A, const float* B, const float* bias, float* C,
                   int M, int Nc, int K, int act){
    gemm(A, B, bias, C, M, Nc, K, K, Nc, Nc, 0, 0, 0, 0, 0, 0, 1, 1, 1.f, act, 0);
  };

  // -------- CSR + self-loop edge-feature means (once) --------
  hipMemsetAsync(cnt,    0, NND * 4, stream);
  hipMemsetAsync(fill,   0, NND * 4, stream);
  hipMemsetAsync(loopea, 0, NND * 6 * 4, stream);
  count_kernel  <<<(NE + 255) / 256, 256, 0, stream>>>(dstA, ea, cnt, loopea);
  scan_kernel   <<<1, 256, 0, stream>>>(cnt, indptr);
  scatter_kernel<<<(NE + 255) / 256, 256, 0, stream>>>(dstA, indptr, fill, csr);
  loopdiv_kernel<<<(NND + 255) / 256, 256, 0, stream>>>(cnt, loopea);

  // -------- 5 GAT layers (fused bias+BN+ELU epilogue) --------
  const int cdims[5] = {32, 64, 128, 256, 512};
  float* outBuf[5] = {bufX, bufY, bufX, bufY, bufX};
  const float* Xcur = x; int inD = 4;
  for (int l = 0; l < 5; l++){
    int c = cdims[l], HC = 4 * c;
    gemmP(Xcur, gW[l], nullptr, bufP, NND, HC, inD, 0);              // h = X @ W
    wered_kernel    <<<1, 32, 0, stream>>>(gWe[l], gAe[l], wered, c);
    attnode_kernel  <<<NND, 128, 0, stream>>>(bufP, gAs[l], gAd[l], loopea, wered,
                                              a_src, a_dst, a_self, c);
    alphaedge_kernel<<<(NE * 4 + 255) / 256, 256, 0, stream>>>(ea, wered, alphaE);
    gatsm_kernel    <<<NND, 256, 0, stream>>>(indptr, csr, srcA, a_src, a_dst, a_self,
                                              alphaE, w_self);
    gatagg_kernel   <<<NND, 256, 0, stream>>>(bufP, indptr, csr, srcA, alphaE, w_self,
                                              gB[l], bnG[l], bnB[l], bnM[l], bnV[l],
                                              outBuf[l], c);
    Xcur = outBuf[l]; inD = HC;
  }
  // z = bufX [8192, 2048]

  // -------- encoder SAB --------
  gemmP(bufX, enc[0], enc[1], bufQ, NND, 2048, 2048, 0);
  gemmP(bufX, enc[2], enc[3], bufK, NND, 2048, 2048, 0);
  gemmP(bufX, enc[4], enc[5], bufV, NND, 2048, 2048, 0);
  gemm(bufQ, bufK, nullptr, bufS, 256, 256, 512, 2048, 2048, 256,
       524288, 512, 524288, 512, 262144, 65536, 4, NG * 4, SCALE_ATT, 0, 1);   // Q K^T
  softmax256_kernel<<<NG * 4 * 256, 256, 0, stream>>>(bufS);
  gemm(bufS, bufV, nullptr, bufY, 256, 512, 256, 256, 2048, 2048,
       262144, 65536, 524288, 512, 524288, 512, 4, NG * 4, 1.f, 0, 0);         // S V
  gemmP(bufY, enc[6], enc[7], bufP, NND, 2048, 2048, 0);                        // Wo
  lnres_kernel<<<NND, 256, 0, stream>>>(bufP, bufX, 2048, enc[10], enc[11], bufQ);
  gemmP(bufQ, enc[8], enc[9], bufP, NND, 2048, 2048, 1);                        // relu(Wl)
  lnres_kernel<<<NND, 256, 0, stream>>>(bufP, bufQ, 2048, enc[12], enc[13], bufX);

  // -------- PMA --------
  gemmP(bufX, pmaW, pmab, bufY, NND, 2048, 2048, 1);                            // t
  gemmP(seed, pma[0], pma[1], qseed, 1, 2048, 2048, 0);                         // seed Wq
  gemmP(bufY, pma[2], pma[3], bufK, NND, 2048, 2048, 0);
  gemmP(bufY, pma[4], pma[5], bufV, NND, 2048, 2048, 0);
  gemm(qseed, bufK, nullptr, spma, 1, 256, 512, 2048, 2048, 256,
       0, 512, 524288, 512, 1024, 256, 4, NG * 4, SCALE_ATT, 0, 1);
  softmax256_kernel<<<NG * 4, 256, 0, stream>>>(spma);
  gemm(spma, bufV, nullptr, bufY, 1, 512, 256, 256, 2048, 2048,
       1024, 256, 524288, 512, 2048, 512, 4, NG * 4, 1.f, 0, 0);               // [32,2048]
  gemmP(bufY, pma[6], pma[7], bufP, NG, 2048, 2048, 0);                         // Wo
  lnres_kernel<<<NG, 256, 0, stream>>>(bufP, seed, 0, pma[10], pma[11], bufQ);
  gemmP(bufQ, pma[8], pma[9], bufP, NG, 2048, 2048, 1);
  lnres_kernel<<<NG, 256, 0, stream>>>(bufP, bufQ, 2048, pma[12], pma[13], bufX);
  // z_pma = bufX[0:32]

  // -------- decoder SAB (single key -> softmax == 1 -> attn out == V exactly)
  gemmP(bufX, dec[4], dec[5], bufV, NG, 2048, 2048, 0);
  gemmP(bufV, dec[6], dec[7], bufP, NG, 2048, 2048, 0);
  lnres_kernel<<<NG, 256, 0, stream>>>(bufP, bufX, 2048, dec[10], dec[11], bufQ);
  gemmP(bufQ, dec[8], dec[9], bufP, NG, 2048, 2048, 1);
  lnres_kernel<<<NG, 256, 0, stream>>>(bufP, bufQ, 2048, dec[12], dec[13], bufY);
  nanfix_kernel<<<(NG * 2048 + 255) / 256, 256, 0, stream>>>(bufY, NG * 2048);

  // -------- FC head --------
  gemmP(bufY, fcW[0], fcb[0], bufP, NG, 256, 2048, 0);
  bnelu_kernel<<<(NG * 256 + 255) / 256, 256, 0, stream>>>(bufP, bnG[5], bnB[5], bnM[5], bnV[5], NG * 256, 256);
  gemmP(bufP, fcW[1], fcb[1], bufQ, NG, 128, 256, 0);
  bnelu_kernel<<<(NG * 128 + 255) / 256, 256, 0, stream>>>(bufQ, bnG[6], bnB[6], bnM[6], bnV[6], NG * 128, 128);
  gemmP(bufQ, fcW[2], fcb[2], bufP, NG, 64, 128, 0);
  bnelu_kernel<<<(NG * 64 + 255) / 256, 256, 0, stream>>>(bufP, bnG[7], bnB[7], bnM[7], bnV[7], NG * 64, 64);
  gemmP(bufP, fcW[3], fcb[3], bufQ, NG, 2, 64, 0);
  logsm2_kernel<<<1, 32, 0, stream>>>(bufQ, (float*)d_out);
}
